// RNN_66692252172317
// MI455X (gfx1250) — matrix operand order repaired and checked by reading, not compiled, not run on hardware
//
#include <hip/hip_runtime.h>
#include <hip/hip_bf16.h>
#include <math.h>

// Problem sizes (match reference)
#define BATCH 128
#define SEQ   512
#define INDIM 512
#define HID   1024
#define KTOT  (HID + INDIM)   // 1536: [h ; x] combined K
#define MTILE 16              // WMMA M = batch tile per workgroup
#define NWAVES 16
#define THREADS (NWAVES * 32)
#define NPW    (HID / NWAVES) // 64 output columns per wave
#define NTILES (NPW / 16)     // 4 WMMA N-tiles per wave

typedef __attribute__((ext_vector_type(16))) __bf16 v16bf;
typedef __attribute__((ext_vector_type(8)))  float  v8f;

struct U4 { unsigned x, y, z, w; };
union ABu { v16bf v; unsigned u[8]; U4 q[2]; };

__device__ __forceinline__ unsigned packbf2(float a, float b) {
  union { __bf16 h[2]; unsigned u; } p;
  p.h[0] = (__bf16)a; p.h[1] = (__bf16)b;
  return p.u;
}

__device__ __forceinline__ float fast_tanh(float z) {
  float e = __expf(2.0f * z);
  return 1.0f - 2.0f / (e + 1.0f);
}

// Build WcombT[k][n] (bf16, row-major K x N):
//   k <  HID : W_lat[n][k]   (so B = W_lat^T for the recurrent GEMM)
//   k >= HID : W_in [n][k-HID]
__global__ void prep_weights(const float* __restrict__ W_in,
                             const float* __restrict__ W_lat,
                             __bf16* __restrict__ WcombT) {
  int k = blockIdx.x;    // 0..KTOT-1
  int n = threadIdx.x;   // 0..HID-1
  float w = (k < HID) ? W_lat[(size_t)n * HID + k]
                      : W_in [(size_t)n * INDIM + (k - HID)];
  WcombT[(size_t)k * HID + n] = (__bf16)w;
}

__global__ void prep_bias(const float* __restrict__ b_in,
                          const float* __restrict__ b_lat,
                          const float* __restrict__ bh,
                          float* __restrict__ biassum) {
  int n = threadIdx.x;
  biassum[n] = b_in[n] + b_lat[n] + bh[n];
}

// Elementwise fp32 -> bf16 conversion of x (fully parallel, off the scan's
// critical path). 8 elements per thread, 16B-aligned stores.
__global__ void prep_x(const float* __restrict__ x, __bf16* __restrict__ xb) {
  size_t i = ((size_t)blockIdx.x * blockDim.x + threadIdx.x) * 8;
  const float4* p = (const float4*)(x + i);
  float4 a = p[0], b = p[1];
  unsigned o[4] = { packbf2(a.x, a.y), packbf2(a.z, a.w),
                    packbf2(b.x, b.y), packbf2(b.z, b.w) };
  *(U4*)(xb + i) = *(const U4*)o;
}

// Persistent scan kernel: one workgroup per 16-batch tile, whole T loop inside.
// XBF = true: x pre-converted to bf16 in workspace (clean dword A loads).
template <bool XBF>
__global__ __launch_bounds__(THREADS)
void rnn_scan(const float* __restrict__ x,      // [B][T][IN] fp32
              const __bf16* __restrict__ xb,    // [B][T][IN] bf16 (if XBF)
              const float* __restrict__ h0,     // [B][HID]   fp32
              const __bf16* __restrict__ WcombT,// [KTOT][HID] bf16
              const float* __restrict__ biassum,// [HID] fp32
              float* __restrict__ out) {        // [B][T][HID] fp32
  __shared__ __bf16 hbuf[2][MTILE * HID];       // double-buffered h tile, 64 KB

  const int tid  = threadIdx.x;
  const int lane = tid & 31;
  const int wave = tid >> 5;
  const int b0   = blockIdx.x * MTILE;

  const int mloc  = lane & 15;       // A-row this lane serves
  const int hi    = lane >> 4;       // lane half (selects K sub-range / M+8)
  const int nloc  = lane & 15;       // C/D column within tile
  const int nbase = wave * NPW;

  // ---- load initial h (fp32 -> bf16) into buffer 0 ----
  for (int i = tid; i < MTILE * HID; i += THREADS) {
    int m = i >> 10, n = i & (HID - 1);
    hbuf[0][i] = (__bf16)h0[(size_t)(b0 + m) * HID + n];
  }
  __syncthreads();

  // Per-lane sub-K offset of each A VGPR's bf16 pair (ISA 16-bit A layout):
  // VGPR v: K = (v<4?0:16) + 8*laneHalf + 2*(v&3) .. +1
  int subK[8];
#pragma unroll
  for (int v = 0; v < 8; ++v)
    subK[v] = ((v < 4) ? 0 : 16) + hi * 8 + 2 * (v & 3);

  float biasv[NTILES];
#pragma unroll
  for (int i = 0; i < NTILES; ++i)
    biasv[i] = biassum[nbase + i * 16 + nloc];

  for (int t = 0; t < SEQ; ++t) {
    const int cur = t & 1, nxt = cur ^ 1;

    // Prefetch next timestep's x tile into cache (global_prefetch_b8):
    // 16 rows x (1 KB bf16 | 2 KB fp32), 128 B granules.
    if (t + 1 < SEQ) {
      if (XBF) {
        if (tid < 128) {
          int m = tid >> 3, c = tid & 7;
          __builtin_prefetch(xb + (size_t)(b0 + m) * SEQ * INDIM +
                                  (size_t)(t + 1) * INDIM + c * 64, 0, 1);
        }
      } else {
        if (tid < 256) {
          int m = tid >> 4, c = tid & 15;
          __builtin_prefetch(x + (size_t)(b0 + m) * SEQ * INDIM +
                                 (size_t)(t + 1) * INDIM + c * 32, 0, 1);
        }
      }
    }

    // C init = broadcast bias (bias depends only on n = column)
    v8f acc[NTILES];
#pragma unroll
    for (int i = 0; i < NTILES; ++i) {
#pragma unroll
      for (int e = 0; e < 8; ++e) acc[i][e] = biasv[i];
    }

    const unsigned short* hrow =
        (const unsigned short*)&hbuf[cur][mloc * HID];
    const float* xrow = x + (size_t)(b0 + mloc) * SEQ * INDIM +
                        (size_t)t * INDIM;
    const unsigned short* xrow16 =
        (const unsigned short*)(xb + (size_t)(b0 + mloc) * SEQ * INDIM +
                                (size_t)t * INDIM);

    // ---- K part 1: recurrent h (from LDS, already bf16) ----
    for (int kk = 0; kk < HID; kk += 32) {
      ABu A;
#pragma unroll
      for (int v = 0; v < 8; ++v)
        A.u[v] = *(const unsigned*)&hrow[kk + subK[v]];
#pragma unroll
      for (int i = 0; i < NTILES; ++i) {
        ABu B;  // lane = N column; K pairs follow subK (same as A)
        const unsigned short* bp = (const unsigned short*)(WcombT + (size_t)kk * HID +
                                   nbase + i * 16 + nloc);
#pragma unroll
        for (int v = 0; v < 8; ++v)
          B.u[v] = (unsigned)bp[(size_t)subK[v] * HID] |
                   ((unsigned)bp[(size_t)(subK[v] + 1) * HID] << 16);
        acc[i] = __builtin_amdgcn_wmma_f32_16x16x32_bf16(
            false, A.v, false, B.v, (short)0, acc[i], false, false);
      }
    }

    // ---- K part 2: input x_t ----
    for (int kk = 0; kk < INDIM; kk += 32) {
      ABu A;
      if (XBF) {
#pragma unroll
        for (int v = 0; v < 8; ++v)
          A.u[v] = *(const unsigned*)&xrow16[kk + subK[v]];
      } else {
#pragma unroll
        for (int v = 0; v < 8; ++v) {
          int k = kk + subK[v];
          A.u[v] = packbf2(xrow[k], xrow[k + 1]);
        }
      }
#pragma unroll
      for (int i = 0; i < NTILES; ++i) {
        ABu B;
        const unsigned short* bp = (const unsigned short*)(WcombT + (size_t)(HID + kk) * HID +
                                   nbase + i * 16 + nloc);
#pragma unroll
        for (int v = 0; v < 8; ++v)
          B.u[v] = (unsigned)bp[(size_t)subK[v] * HID] |
                   ((unsigned)bp[(size_t)(subK[v] + 1) * HID] << 16);
        acc[i] = __builtin_amdgcn_wmma_f32_16x16x32_bf16(
            false, A.v, false, B.v, (short)0, acc[i], false, false);
      }
    }

    // ---- activation + writeback (C/D layout: lane->n, VGPR v -> m=v+8*hi) ----
#pragma unroll
    for (int i = 0; i < NTILES; ++i) {
      int n = nbase + i * 16 + nloc;
#pragma unroll
      for (int v = 0; v < 8; ++v) {
        int m = v + hi * 8;
        float th = fast_tanh(acc[i][v]);
        out[(size_t)(b0 + m) * SEQ * HID + (size_t)t * HID + n] = th;
        hbuf[nxt][m * HID + n] = (__bf16)th;
      }
    }
    __syncthreads();  // next step reads hbuf[nxt]; also protects hbuf[cur] reuse
  }
}

extern "C" void kernel_launch(void* const* d_in, const int* in_sizes, int n_in,
                              void* d_out, int out_size, void* d_ws, size_t ws_size,
                              hipStream_t stream) {
  const float* x     = (const float*)d_in[0];
  const float* h0    = (const float*)d_in[1];
  const float* W_in  = (const float*)d_in[2];
  const float* b_in  = (const float*)d_in[3];
  const float* W_lat = (const float*)d_in[4];
  const float* b_lat = (const float*)d_in[5];
  const float* bh    = (const float*)d_in[6];

  // workspace layout:
  //   [0, 3 MB)            bf16 WcombT
  //   [+0, 4 KB)           fp32 biassum
  //   [+4 KB, +64 MB)      bf16 x (optional, if ws_size permits)
  const size_t off_bias = (size_t)KTOT * HID * sizeof(__bf16);   // 3,145,728
  const size_t off_x    = off_bias + 4096;
  const size_t x_elems  = (size_t)BATCH * SEQ * INDIM;
  const size_t need_xbf = off_x + x_elems * sizeof(__bf16);

  __bf16* WcombT = (__bf16*)d_ws;
  float*  biassum = (float*)((char*)d_ws + off_bias);
  __bf16* xb      = (__bf16*)((char*)d_ws + off_x);

  const bool use_xbf = (ws_size >= need_xbf);

  prep_weights<<<KTOT, HID, 0, stream>>>(W_in, W_lat, WcombT);
  prep_bias<<<1, HID, 0, stream>>>(b_in, b_lat, bh, biassum);

  if (use_xbf) {
    prep_x<<<(int)(x_elems / (256 * 8)), 256, 0, stream>>>(x, xb);
    rnn_scan<true><<<BATCH / MTILE, THREADS, 0, stream>>>(
        x, xb, h0, WcombT, biassum, (float*)d_out);
  } else {
    rnn_scan<false><<<BATCH / MTILE, THREADS, 0, stream>>>(
        x, xb, h0, WcombT, biassum, (float*)d_out);
  }
}
